// TreeGetter_59639915872743
// MI455X (gfx1250) — matrix operand order repaired and checked by reading, not compiled, not run on hardware
//
#include <hip/hip_runtime.h>
#include <math.h>

typedef __bf16 bf16;
typedef __attribute__((ext_vector_type(16))) __bf16 v16bf;
typedef __attribute__((ext_vector_type(8)))  float  v8f;

#define CDIM 768
#define KTOT 2304   // 3 * 768 (im2col K)
#define NT   (KTOT / 32)   // 72 K-steps
#define BM 128      // output positions per block tile
#define BN 64       // output channels per block tile
#define BK 32       // K step = one bf16 WMMA K
#define LDA 40      // LDS row stride (bf16) for A tile (80B rows, 16B aligned)
#define LDB 40      // LDS row stride (bf16) for B tile (n-major, k contiguous)
#define BZ 16
#define PMAX 2048

__device__ __forceinline__ int halve_len(int len, int times) {
  for (int i = 0; i < times; ++i) len = (len - 1) / 2 + 1;
  return len;
}

__device__ __forceinline__ float gelu_exact(float x) {
  return 0.5f * x * (1.0f + erff(x * 0.70710678118654752440f));
}

__device__ __forceinline__ unsigned lds_off(const void* p) {
  // generic LDS pointer = aperture<<32 | lds_byte_offset -> truncate
  return (unsigned)(uintptr_t)p;
}

// 16B global -> LDS async copy (gfx1250 VGLOBAL async path, tracked by ASYNCcnt)
__device__ __forceinline__ void async_b128(unsigned lds, const void* g) {
  asm volatile("global_load_async_to_lds_b128 %0, %1, off"
               :: "v"(lds), "v"(g) : "memory");
}
__device__ __forceinline__ void wait_async_le3() {
  asm volatile("s_wait_asynccnt 0x3" ::: "memory");
}
__device__ __forceinline__ void wait_async_le0() {
  asm volatile("s_wait_asynccnt 0x0" ::: "memory");
}

// One-time: w[O=768][I=768][Kw=3] f32 -> wBt[n][k*768 + i] bf16 (n-major, GEMM-K contiguous)
__global__ void prep_weights(const float* __restrict__ w, bf16* __restrict__ wBt) {
  size_t idx = (size_t)blockIdx.x * blockDim.x + threadIdx.x;
  if (idx >= (size_t)CDIM * KTOT) return;
  int n  = (int)(idx / KTOT);
  int kk = (int)(idx % KTOT);
  int k  = kk / CDIM;
  int i  = kk % CDIM;
  wBt[idx] = (bf16)w[(size_t)n * KTOT + i * 3 + k];
}

// Level-1 input: f32 [BZ][2048][768] -> masked bf16 padded [BZ][2050][768]
// (row 0 and row 2049 zero; row p+1 = x0[p] * (p < len))
__global__ void convert_x0(const float* __restrict__ x0, const int* __restrict__ lengths,
                           bf16* __restrict__ xpad) {
  const size_t stride = (size_t)(PMAX + 2) * CDIM;
  size_t idx = ((size_t)blockIdx.x * blockDim.x + threadIdx.x) * 4;
  if (idx >= (size_t)BZ * stride) return;
  int b = (int)(idx / stride);
  size_t rem = idx % stride;
  int r = (int)(rem / CDIM);
  int c = (int)(rem % CDIM);
  int p = r - 1;
  union { bf16 h[4]; uint2 u; } o;
  if (p >= 0 && p < PMAX && p < lengths[b]) {
    const float* s = x0 + ((size_t)b * PMAX + p) * CDIM + c;
    o.h[0] = (bf16)s[0]; o.h[1] = (bf16)s[1]; o.h[2] = (bf16)s[2]; o.h[3] = (bf16)s[3];
  } else {
    o.u = make_uint2(0u, 0u);
  }
  *(uint2*)(xpad + idx) = o.u;
}

// Zero pad rows 0 and L+1 of buf[BZ][L+2][768] (stride changes per level).
__global__ void zero_pads(bf16* __restrict__ buf, int L) {
  const int per = 2 * CDIM / 4;  // vec4 slots per sample
  int idx = blockIdx.x * blockDim.x + threadIdx.x;
  if (idx >= BZ * per) return;
  int b = idx / per, r4 = idx % per;
  int top = r4 / (CDIM / 4);
  int c = (r4 % (CDIM / 4)) * 4;
  size_t row = top ? (size_t)(L + 1) : 0;
  *(uint2*)(buf + ((size_t)b * (L + 2) + row) * CDIM + c) = make_uint2(0u, 0u);
}

// Writes all 11 per-level masks into d_out at their interleaved offsets.
__global__ void masks_kernel(const int* __restrict__ lengths, float* __restrict__ out) {
  const int level = blockIdx.x + 1;  // 1..11
  size_t off = 0;
  int P = PMAX;
  for (int j = 1; j <= level; ++j) {
    P >>= 1;
    if (j < level) off += (size_t)BZ * P * CDIM + (size_t)BZ * P;
    else           off += (size_t)BZ * P * CDIM;
  }
  const int total = BZ * P;
  for (int i = threadIdx.x; i < total; i += blockDim.x) {
    int b = i / P, p = i % P;
    int len = halve_len(lengths[b], level);
    out[off + i] = (p < len) ? 1.0f : 0.0f;
  }
}

// Conv-as-GEMM, bf16 WMMA w/ f32 accumulate, async-to-LDS double buffering.
// src is zero-padded & pre-masked: [BZ][Pin+2][768] bf16, data rows at p+1.
// CONV1 (stride 1): +bias, exact GELU, mask p<len_in, store bf16 h (padded).
// !CONV1 (stride 2): +bias, mask p<len_out, store f32 feas slice + bf16 x_next (padded).
template <int STRIDE, bool CONV1>
__global__ __launch_bounds__(256) void conv_gemm_kernel(
    const bf16* __restrict__ src, const bf16* __restrict__ wBt,
    const float* __restrict__ bias, const int* __restrict__ lengths,
    int Pin, int Pout, int halvings_in,
    bf16* __restrict__ dst_bf, float* __restrict__ dst_f) {
  __shared__ bf16 As[2][BM * LDA];
  __shared__ bf16 Bs[2][BN * LDB];

  const int tid  = threadIdx.x;
  const int b    = blockIdx.z;
  const int m0   = blockIdx.x * BM;
  const int n0   = blockIdx.y * BN;
  const int Pcur = CONV1 ? Pin : Pout;

  const int len_in  = halve_len(lengths[b], halvings_in);
  const int len_out = (len_in - 1) / 2 + 1;

  const int lane  = tid & 31;
  const int wave  = tid >> 5;
  const int wm    = wave & 3;
  const int wn    = wave >> 2;
  const int mlane = lane & 15;
  const int hi    = lane >> 4;

  // A staging role: 2 threads/row, 16 bf16 (2x16B) each. B: 4 threads/row, 8 bf16 (16B).
  const int ar    = tid >> 1;
  const int ahalf = (tid & 1) * 16;
  const int brow  = tid >> 2;
  const int bq    = (tid & 3) * 8;

  // Per-tap A base pointers (loop-invariant; clamp lands in zero pad rows).
  const bf16* abase[3];
#pragma unroll
  for (int s = 0; s < 3; ++s) {
    int ip = (m0 + ar) * STRIDE - 1 + s;      // >= -1 always
    ip = ip > Pin ? Pin : ip;                 // rows beyond data -> top pad (zeros)
    abase[s] = src + ((size_t)b * (Pin + 2) + (ip + 1)) * CDIM + ahalf;
  }

  const unsigned ldsA[2] = { lds_off(&As[0][ar * LDA + ahalf]),
                             lds_off(&As[1][ar * LDA + ahalf]) };
  const unsigned ldsB[2] = { lds_off(&Bs[0][brow * LDB + bq]),
                             lds_off(&Bs[1][brow * LDB + bq]) };

  // Issue-side state (runs 2 tiles ahead of compute).
  const bf16* iaptr = abase[0];
  const bf16* ibptr = wBt + (size_t)(n0 + brow) * KTOT + bq;
  int icb = 0, ishift = 0;

  auto issue_tile = [&](int buf) {
    unsigned la = ldsA[buf], lb = ldsB[buf];
    async_b128(la, iaptr);
    async_b128(la + 16u, iaptr + 8);
    async_b128(lb, ibptr);
    iaptr += BK;
    ibptr += BK;
    if (++icb == KTOT / CDIM * 8) { /* no-op placeholder; true at each tap end */ }
    if (icb == CDIM / BK) {  // 24 channel-blocks per tap
      icb = 0;
      if (ishift < 2) iaptr = abase[++ishift];
    }
  };

  const v8f vzero = {0.f, 0.f, 0.f, 0.f, 0.f, 0.f, 0.f, 0.f};
  v8f acc[2][2];
  acc[0][0] = vzero; acc[0][1] = vzero; acc[1][0] = vzero; acc[1][1] = vzero;

  auto compute = [&](int cur) {
    v16bf afrag[2], bfrag[2];
#pragma unroll
    for (int tm = 0; tm < 2; ++tm) {
      const bf16* arow = &As[cur][(wm * 32 + tm * 16 + mlane) * LDA];
#pragma unroll
      for (int j = 0; j < 8; ++j) {
        // A 16x32 bf16: lanes 0-15 V0-3 = K0..7, V4-7 = K16..23; lanes 16-31 offset +8
        int kk = ((j < 4) ? 0 : 16) + hi * 8 + 2 * (j & 3);
        afrag[tm][2 * j]     = arow[kk];
        afrag[tm][2 * j + 1] = arow[kk + 1];
      }
    }
#pragma unroll
    for (int tn = 0; tn < 2; ++tn) {
      const bf16* brw = &Bs[cur][(wn * 32 + tn * 16 + mlane) * LDB];
#pragma unroll
      for (int j = 0; j < 8; ++j) {
        // B 32x16 bf16: lanes 0-15 V0-3 = K0..7, V4-7 = K16..23; lanes 16-31 offset +8; N = lane%16
        int kk = ((j < 4) ? 0 : 16) + hi * 8 + 2 * (j & 3);
        bfrag[tn][2 * j]     = brw[kk];
        bfrag[tn][2 * j + 1] = brw[kk + 1];
      }
    }
#pragma unroll
    for (int tm = 0; tm < 2; ++tm)
#pragma unroll
      for (int tn = 0; tn < 2; ++tn)
        acc[tm][tn] = __builtin_amdgcn_wmma_f32_16x16x32_bf16(
            false, afrag[tm], false, bfrag[tn], (short)0, acc[tm][tn], false, false);
  };

  // Pipeline: prefetch 2 tiles, steady state keeps one tile in flight during WMMAs.
  issue_tile(0);
  issue_tile(1);
  for (int kb = 0; kb < NT - 1; ++kb) {
    const int cur = kb & 1;
    wait_async_le3();   // my 3 ops for tile kb done (tile kb+1 may be in flight)
    __syncthreads();    // all waves' tile-kb LDS writes visible
    compute(cur);
    __syncthreads();    // all waves done reading buf[cur]
    if (kb + 2 < NT) issue_tile(cur);
  }
  wait_async_le0();
  __syncthreads();
  compute((NT - 1) & 1);

  // Epilogue (C/D f32 16x16: row = v + 8*hi, col = lane%16)
#pragma unroll
  for (int tn = 0; tn < 2; ++tn) {
    const int n  = n0 + wn * 32 + tn * 16 + mlane;
    const float bv = bias[n];
#pragma unroll
    for (int tm = 0; tm < 2; ++tm) {
#pragma unroll
      for (int v = 0; v < 8; ++v) {
        const int p = m0 + wm * 32 + tm * 16 + v + 8 * hi;
        if (p >= Pcur) continue;
        float val = acc[tm][tn][v] + bv;
        if (CONV1) {
          val = gelu_exact(val);
          if (p >= len_in) val = 0.0f;  // m1 mask
          dst_bf[((size_t)b * (Pin + 2) + p + 1) * CDIM + n] = (bf16)val;
        } else {
          if (p >= len_out) val = 0.0f;  // m2 mask
          dst_f[((size_t)b * Pout + p) * CDIM + n] = val;
          dst_bf[((size_t)b * (Pout + 2) + p + 1) * CDIM + n] = (bf16)val;
        }
      }
    }
  }
}

extern "C" void kernel_launch(void* const* d_in, const int* in_sizes, int n_in,
                              void* d_out, int out_size, void* d_ws, size_t ws_size,
                              hipStream_t stream) {
  const float* x0      = (const float*)d_in[0];
  const int*   lengths = (const int*)d_in[1];
  const float* w1      = (const float*)d_in[2];
  const float* b1      = (const float*)d_in[3];
  const float* w2      = (const float*)d_in[4];
  const float* b2      = (const float*)d_in[5];
  float* out = (float*)d_out;

  char* ws = (char*)d_ws;
  bf16* wBt1 = (bf16*)ws; ws += (size_t)CDIM * KTOT * sizeof(bf16);
  bf16* wBt2 = (bf16*)ws; ws += (size_t)CDIM * KTOT * sizeof(bf16);
  bf16* xpad = (bf16*)ws; ws += (size_t)BZ * (PMAX + 2) * CDIM * sizeof(bf16);  // level-1 x
  bf16* hpad = (bf16*)ws; ws += (size_t)BZ * (PMAX + 2) * CDIM * sizeof(bf16);  // h
  bf16* xbuf = (bf16*)ws;  // BZ * (1024+2) * 768 bf16, next-level x

  {
    const int totalW = CDIM * KTOT;
    prep_weights<<<(totalW + 255) / 256, 256, 0, stream>>>(w1, wBt1);
    prep_weights<<<(totalW + 255) / 256, 256, 0, stream>>>(w2, wBt2);
    const size_t total4 = (size_t)BZ * (PMAX + 2) * CDIM / 4;
    convert_x0<<<(unsigned)((total4 + 255) / 256), 256, 0, stream>>>(x0, lengths, xpad);
  }
  masks_kernel<<<11, 256, 0, stream>>>(lengths, out);

  const int padElems = (BZ * 2 * CDIM / 4 + 255) / 256;
  size_t off = 0;
  int Pin = PMAX;
  for (int level = 1; level <= 11; ++level) {
    const int Pout = (Pin - 1) / 2 + 1;
    const bf16* srcX = (level == 1) ? xpad : xbuf;

    zero_pads<<<padElems, 256, 0, stream>>>(hpad, Pin);
    dim3 g1((Pin + BM - 1) / BM, CDIM / BN, BZ);
    conv_gemm_kernel<1, true><<<g1, 256, 0, stream>>>(
        srcX, wBt1, b1, lengths, Pin, Pout, level - 1, hpad, nullptr);

    zero_pads<<<padElems, 256, 0, stream>>>(xbuf, Pout);
    dim3 g2((Pout + BM - 1) / BM, CDIM / BN, BZ);
    conv_gemm_kernel<2, false><<<g2, 256, 0, stream>>>(
        hpad, wBt2, b2, lengths, Pin, Pout, level - 1, xbuf, out + off);

    off += (size_t)BZ * Pout * CDIM + (size_t)BZ * Pout;
    Pin = Pout;
  }
}
